// VQGanTokenizer_50723563766041
// MI455X (gfx1250) — matrix operand order repaired and checked by reading, not compiled, not run on hardware
//
#include <hip/hip_runtime.h>
#include <hip/hip_bf16.h>

// ---------------------------------------------------------------------------
// CDNA5 (gfx1250) VQGAN encoder + vector quantizer, wave32 + WMMA bf16.
// All GEMM-like math (convs as implicit GEMM, attention, VQ distances) runs
// through v_wmma_f32_16x16x32_bf16 (f32 accumulate).  Conv weights are staged
// per-block into LDS as bf16 (8x reuse), A fragments read via 16B LDS loads.
// ---------------------------------------------------------------------------

typedef __attribute__((ext_vector_type(16))) __bf16 v16bf;
typedef __attribute__((ext_vector_type(8)))  __bf16 v8bf;
typedef __attribute__((ext_vector_type(8)))  float  v8f;

__device__ __forceinline__ __bf16 f2bf(float f) {
  unsigned u = __builtin_bit_cast(unsigned, f);
  u = (u + 0x7FFFu + ((u >> 16) & 1u)) >> 16;          // round-to-nearest-even
  unsigned short h = (unsigned short)u;
  return __builtin_bit_cast(__bf16, h);
}

// 16-bit A-matrix (16x32) K index for element e (0..15) given lane-half.
// ISA 7.12.2: per lane the A fragment is two contiguous 8-wide K runs:
//   e in [0,8):  k = 8*half + e
//   e in [8,16): k = 16 + 8*half + (e-8)
__device__ __forceinline__ int a_frag_k(int e, int half) {
  return (e < 8) ? ((half << 3) + e) : (16 + (half << 3) + (e - 8));
}

__device__ __forceinline__ v8f wmma_bf16(v16bf a, v16bf b, v8f c) {
  return __builtin_amdgcn_wmma_f32_16x16x32_bf16(false, a, false, b,
                                                 (short)0, c, false, false);
}

// Pack 4 float4 (16 consecutive-in-K floats per the fragment layout).
__device__ __forceinline__ v16bf pack16(float4 a0, float4 a1, float4 a2,
                                        float4 a3) {
  v16bf f;
  f[0]  = f2bf(a0.x); f[1]  = f2bf(a0.y); f[2]  = f2bf(a0.z); f[3]  = f2bf(a0.w);
  f[4]  = f2bf(a1.x); f[5]  = f2bf(a1.y); f[6]  = f2bf(a1.z); f[7]  = f2bf(a1.w);
  f[8]  = f2bf(a2.x); f[9]  = f2bf(a2.y); f[10] = f2bf(a2.z); f[11] = f2bf(a2.w);
  f[12] = f2bf(a3.x); f[13] = f2bf(a3.y); f[14] = f2bf(a3.z); f[15] = f2bf(a3.w);
  return f;
}

// ---------------------------------------------------------------------------
// Implicit-GEMM convolution, templated on kernel size (3x3 / 1x1 => all index
// math is constant mul/shift).  M = Cout (64 per wave: 4 sub-tiles), N =
// N*Hout*Wout (tile 16), K = Cin*KH*KW stepped by 32.
//  - All 8 waves of a block share one Cout-tile: the 64x32 weight slab is
//    cooperatively staged into LDS as bf16 once per K-step (8x HBM reuse),
//    and each wave builds A fragments with two 16B LDS vector reads.
//  - B (im2col) is gathered once per K-step and reused by 4 WMMAs.
//  - Stores coalesced: lanes store consecutive spatial positions.
// OOB input reads return 0 => pad=1 convs and the bottom/right-only pad of
// downsample (stride=2, pad=0) fall out naturally.
// ---------------------------------------------------------------------------
template <int KH, int KW>
__global__ void conv_wmma_kernel(const float* __restrict__ x,
                                 const float* __restrict__ w,
                                 const float* __restrict__ bias,
                                 float* __restrict__ y,
                                 int N, int Cin, int Hin, int Win,
                                 int Cout, int Hout, int Wout,
                                 int stride, int pad) {
  constexpr int KHW = KH * KW;
  const int tid  = threadIdx.x;
  const int wave = tid >> 5;
  const int lane = tid & 31;
  const int tilesN8 = (N * Hout * Wout) >> 7;      // groups of 8 spatial tiles
  const int tileM = blockIdx.x / tilesN8;          // shared by whole block
  const int tileN = (blockIdx.x - tileM * tilesN8) * 8 + wave;

  const int half = lane >> 4;
  const int lm   = lane & 15;
  const int Ktot = Cin * KHW;
  const int cobase = tileM * 64;

  // decode output position (B-side column / store column), n = lm in tile
  const int p   = tileN * 16 + lm;
  const int ow  = p % Wout;
  const int t0  = p / Wout;
  const int oh  = t0 % Hout;
  const int nim = t0 / Hout;

  __shared__ __align__(16) __bf16 wlds[64 * 32];   // 64 co x 32 k, bf16

  v8f acc[4] = {{}, {}, {}, {}};
  for (int k0 = 0; k0 < Ktot; k0 += 32) {
    // ---- cooperative weight slab stage: global f32 -> LDS bf16 ----
    __syncthreads();                               // previous readers done
#pragma unroll
    for (int j = 0; j < 8; ++j) {
      int i  = tid + (j << 8);                     // 2048 elems / 256 threads
      int cl = i >> 5, kk = i & 31;
      float wv = 0.0f;
      if (k0 + kk < Ktot) {
        const float* gp = w + (long)(cobase + cl) * Ktot + k0 + kk;
        wv = *gp;
        if (j == 0) __builtin_prefetch(gp + 32, 0, 3);  // next K slab
      }
      wlds[cl * 32 + kk] = f2bf(wv);
    }
    __syncthreads();

    // ---- B = im2col(x): gathered once, reused by 4 WMMAs ----
    v16bf bf;
#pragma unroll
    for (int e = 0; e < 16; ++e) {
      int kb = k0 + a_frag_k(e, half);
      float bv = 0.0f;
      if (kb < Ktot) {
        int ci = kb / KHW, r = kb - ci * KHW;
        int kh = r / KW,   kw = r - kh * KW;
        int hin = oh * stride + kh - pad;
        int win = ow * stride + kw - pad;
        if ((unsigned)hin < (unsigned)Hin && (unsigned)win < (unsigned)Win)
          bv = x[((nim * Cin + ci) * Hin + hin) * Win + win];
      }
      bf[e] = f2bf(bv);
    }

    // ---- A fragments from LDS: two contiguous 8-elem K runs per lane ----
#pragma unroll
    for (int mt = 0; mt < 4; ++mt) {
      const v8bf* ar = (const v8bf*)(wlds + (mt * 16 + lm) * 32 + (half << 3));
      v8bf lo = ar[0];                             // k = 8h .. 8h+7
      v8bf hi = ar[2];                             // k = 16+8h .. 16+8h+7
      v16bf af = __builtin_shufflevector(lo, hi, 0, 1, 2, 3, 4, 5, 6, 7,
                                         8, 9, 10, 11, 12, 13, 14, 15);
      acc[mt] = wmma_bf16(af, bf, acc[mt]);
    }
  }
  // C/D layout: element r -> m = r + 8*half, n = lm  (coalesced over lanes)
#pragma unroll
  for (int mt = 0; mt < 4; ++mt) {
#pragma unroll
    for (int r = 0; r < 8; ++r) {
      int co = cobase + mt * 16 + r + 8 * half;
      y[((nim * Cout + co) * Hout + oh) * Wout + ow] = acc[mt][r] + bias[co];
    }
  }
}

// ---------------------------------------------------------------------------
// GroupNorm (32 groups) with optional fused SiLU. One block per (batch,group).
// ---------------------------------------------------------------------------
__global__ void groupnorm_kernel(const float* __restrict__ in,
                                 float* __restrict__ out,
                                 const float* __restrict__ gg,
                                 const float* __restrict__ bb,
                                 int C, int HW, int dosilu) {
  const int n   = blockIdx.x >> 5;
  const int grp = blockIdx.x & 31;
  const int cpg = C >> 5;
  const long base = ((long)n * C + (long)grp * cpg) * HW;
  const int cnt = cpg * HW;

  float s = 0.f, ss = 0.f;
  for (int i = threadIdx.x; i < cnt; i += 256) {
    float v = in[base + i];
    s += v; ss += v * v;
  }
  __shared__ float rs[256], rq[256];
  rs[threadIdx.x] = s; rq[threadIdx.x] = ss;
  __syncthreads();
  for (int off = 128; off; off >>= 1) {
    if (threadIdx.x < off) {
      rs[threadIdx.x] += rs[threadIdx.x + off];
      rq[threadIdx.x] += rq[threadIdx.x + off];
    }
    __syncthreads();
  }
  const float mean = rs[0] / (float)cnt;
  const float var  = rq[0] / (float)cnt - mean * mean;
  const float inv  = rsqrtf(var + 1e-5f);
  for (int i = threadIdx.x; i < cnt; i += 256) {
    int c = grp * cpg + i / HW;
    float v = (in[base + i] - mean) * inv * gg[c] + bb[c];
    if (dosilu) v = v / (1.f + expf(-v));
    out[base + i] = v;
  }
}

__global__ void add_kernel(const float* __restrict__ a,
                           const float* __restrict__ b,
                           float* __restrict__ o, int n) {
  int i = blockIdx.x * blockDim.x + threadIdx.x;
  if (i < n) o[i] = a[i] + b[i];
}

// ---------------------------------------------------------------------------
// Attention: scores[b,i,j] = scale * sum_c q[b,c,i]*k[b,c,j]   (M=i,N=j,K=c)
// Both operands K-strided (stride L) -> gathered; tiny share of total FLOPs.
// ---------------------------------------------------------------------------
__global__ void attn_scores_kernel(const float* __restrict__ q,
                                   const float* __restrict__ k,
                                   float* __restrict__ s,
                                   int B, int C, int L, float scale) {
  const int wid  = (blockIdx.x * blockDim.x + threadIdx.x) >> 5;
  const int lane = threadIdx.x & 31;
  const int tiles = L >> 4;
  if (wid >= B * tiles * tiles) return;
  const int b   = wid / (tiles * tiles);
  const int rem = wid - b * tiles * tiles;
  const int ti = rem / tiles, tj = rem - (rem / tiles) * tiles;
  const int half = lane >> 4, lm = lane & 15;

  v8f acc = {};
  for (int k0 = 0; k0 < C; k0 += 32) {
    v16bf af, bf;
#pragma unroll
    for (int e = 0; e < 16; ++e) {
      int ka = k0 + a_frag_k(e, half);
      af[e] = f2bf(q[(b * C + ka) * L + ti * 16 + lm]);
      int kb = k0 + a_frag_k(e, half);
      bf[e] = f2bf(k[(b * C + kb) * L + tj * 16 + lm]);
    }
    acc = wmma_bf16(af, bf, acc);
  }
#pragma unroll
  for (int r = 0; r < 8; ++r) {
    int i = ti * 16 + r + 8 * half, j = tj * 16 + lm;
    s[(b * L + i) * L + j] = acc[r] * scale;
  }
}

// out[b,c,i] = sum_j v[b,c,j] * a[b,i,j]   (M=c, N=i, K=j)
// Both operands contiguous in K => all-vector fragment loads.
__global__ void attn_out_kernel(const float* __restrict__ v,
                                const float* __restrict__ a,
                                float* __restrict__ out,
                                int B, int C, int L) {
  const int wid  = (blockIdx.x * blockDim.x + threadIdx.x) >> 5;
  const int lane = threadIdx.x & 31;
  const int tc_n = C >> 4, ti_n = L >> 4;
  if (wid >= B * tc_n * ti_n) return;
  const int b   = wid / (tc_n * ti_n);
  const int rem = wid - b * tc_n * ti_n;
  const int tc = rem / ti_n, ti = rem - tc * ti_n;
  const int half = lane >> 4, lm = lane & 15;

  const float* vr0 = v + ((long)b * C + tc * 16 + lm) * L;   // A row (m=c)
  const float* ar0 = a + ((long)b * L + ti * 16 + lm) * L;   // B row (n=i)

  v8f acc = {};
  for (int k0 = 0; k0 < L; k0 += 32) {
    const float* vr = vr0 + k0;
    v16bf af = pack16(*(const float4*)(vr + (half << 3)),
                      *(const float4*)(vr + (half << 3) + 4),
                      *(const float4*)(vr + 16 + (half << 3)),
                      *(const float4*)(vr + 16 + (half << 3) + 4));
    const float4* br = (const float4*)(ar0 + k0 + (half << 3));
    v16bf bf = pack16(br[0], br[1], br[4], br[5]);
    acc = wmma_bf16(af, bf, acc);
  }
#pragma unroll
  for (int r = 0; r < 8; ++r) {
    int c = tc * 16 + r + 8 * half, i = ti * 16 + lm;
    out[((long)b * C + c) * L + i] = acc[r];
  }
}

// Row softmax over L=256 (blockDim == L)
__global__ void softmax_kernel(float* __restrict__ s, int L) {
  const int row = blockIdx.x, t = threadIdx.x;
  float v = s[row * L + t];
  __shared__ float red[256];
  red[t] = v; __syncthreads();
  for (int off = 128; off; off >>= 1) {
    if (t < off) red[t] = fmaxf(red[t], red[t + off]);
    __syncthreads();
  }
  float m = red[0]; __syncthreads();
  float e = expf(v - m);
  red[t] = e; __syncthreads();
  for (int off = 128; off; off >>= 1) {
    if (t < off) red[t] += red[t + off];
    __syncthreads();
  }
  s[row * L + t] = e / red[0];
}

// ---------------------------------------------------------------------------
// Vector quantization.  argmin_n ( |e_n|^2 - 2 z.e_n )  — the |z|^2 term is
// row-constant and dropped.  One wave per 16 z-rows; A (z rows) preloaded for
// the full K=256; codebook B fragments are contiguous => vector loads.
// ---------------------------------------------------------------------------
__global__ void cnorm_kernel(const float* __restrict__ cb,
                             float* __restrict__ cn) {
  int i = blockIdx.x * blockDim.x + threadIdx.x;
  if (i < 16384) {
    float s = 0.f;
    for (int d = 0; d < 256; ++d) { float v = cb[i * 256 + d]; s += v * v; }
    cn[i] = s;
  }
}

__global__ void vq_argmin_kernel(const float* __restrict__ h,   // [8,256,16,16]
                                 const float* __restrict__ cb,  // [16384,256]
                                 const float* __restrict__ cn,  // [16384]
                                 int* __restrict__ idx_out) {   // [2048]
  const int tile = blockIdx.x;          // 128 tiles of 16 rows
  const int lane = threadIdx.x;
  const int half = lane >> 4, lm = lane & 15;
  const int row = tile * 16 + lm;
  const int b = row >> 8, s = row & 255;

  v16bf afrag[8];
#pragma unroll
  for (int kk = 0; kk < 8; ++kk) {
#pragma unroll
    for (int e = 0; e < 16; ++e) {
      int d = kk * 32 + a_frag_k(e, half);
      afrag[kk][e] = f2bf(h[(b * 256 + d) * 256 + s]);
    }
  }

  float minv[8]; int mini[8];
#pragma unroll
  for (int r = 0; r < 8; ++r) { minv[r] = 3.4e38f; mini[r] = 0; }

  for (int nt = 0; nt < 1024; ++nt) {
    const int code = nt * 16 + lm;
    const float* crow = cb + (long)code * 256 + (half << 3);
    v8f acc = {};
#pragma unroll
    for (int kk = 0; kk < 8; ++kk) {
      const float4* cr = (const float4*)(crow + kk * 32);
      v16bf bf = pack16(cr[0], cr[1], cr[4], cr[5]);
      acc = wmma_bf16(afrag[kk], bf, acc);
    }
    const float c2 = cn[code];
#pragma unroll
    for (int r = 0; r < 8; ++r) {
      float d = c2 - 2.f * acc[r];
      if (d < minv[r]) { minv[r] = d; mini[r] = code; }  // strict < keeps first
    }
  }
  // reduce across the 16 lanes of each half (tie-break: smaller index)
#pragma unroll
  for (int r = 0; r < 8; ++r) {
    for (int off = 8; off > 0; off >>= 1) {
      float ov = __shfl_xor(minv[r], off, 16);
      int   oi = __shfl_xor(mini[r], off, 16);
      if (ov < minv[r] || (ov == minv[r] && oi < mini[r])) {
        minv[r] = ov; mini[r] = oi;
      }
    }
    if (lm == 0) idx_out[tile * 16 + r + 8 * half] = mini[r];
  }
}

// quant = codebook[idx] (straight-through forward value), partial loss sums,
// indices to output.  One block per z-row (blockDim == 256 == D).
__global__ void vq_finalize_kernel(const float* __restrict__ h,
                                   const float* __restrict__ cb,
                                   const int* __restrict__ idx,
                                   float* __restrict__ quant,
                                   float* __restrict__ partial, // [2048]
                                   int* __restrict__ idx_out) {
  const int p = blockIdx.x, d = threadIdx.x;
  const int b = p >> 8, s = p & 255;
  const int id = idx[p];
  const float z  = h[(b * 256 + d) * 256 + s];
  const float zq = cb[id * 256 + d];
  quant[(b * 256 + d) * 256 + s] = zq;
  const float df = zq - z;
  __shared__ float red[256];
  red[d] = df * df; __syncthreads();
  for (int off = 128; off; off >>= 1) {
    if (d < off) red[d] += red[d + off];
    __syncthreads();
  }
  if (d == 0) { partial[p] = red[0]; idx_out[p] = id; }
}

__global__ void loss_reduce_kernel(const float* __restrict__ partial,
                                   float* __restrict__ loss) {
  __shared__ float red[256];
  float s = 0.f;
  for (int i = threadIdx.x; i < 2048; i += 256) s += partial[i];
  red[threadIdx.x] = s; __syncthreads();
  for (int off = 128; off; off >>= 1) {
    if (threadIdx.x < off) red[threadIdx.x] += red[threadIdx.x + off];
    __syncthreads();
  }
  if (threadIdx.x == 0) *loss = red[0] * (1.25f / 524288.0f); // (1+BETA)*mean
}

// ---------------------------------------------------------------------------
// Host orchestration.  Parameter leaves follow JAX pytree flattening: dict
// keys sorted alphabetically, lists in order.  148 leaves total; x is last.
// ---------------------------------------------------------------------------
extern "C" void kernel_launch(void* const* d_in, const int* in_sizes, int n_in,
                              void* d_out, int out_size, void* d_ws,
                              size_t ws_size, hipStream_t stream) {
  (void)in_sizes; (void)n_in; (void)out_size; (void)ws_size;
  auto P = [&](int i) -> const float* { return (const float*)d_in[i]; };

  float* WS = (float*)d_ws;
  const size_t CAP = 67108864ull;           // 8*128*256*256 floats
  float* X  = WS;
  float* T1 = WS + CAP;
  float* T2 = WS + 2 * CAP;
  float* EX = WS + 3 * CAP;
  float* Q   = EX;                          // 8*512*256
  float* Kt  = EX + 1048576;
  float* V   = EX + 2097152;
  float* S   = EX + 3145728;                // 8*256*256
  float* CN  = EX + 3670016;                // 16384
  int*   IDX = (int*)(EX + 3686400);        // 2048
  float* LS  = EX + 3688448;                // 2048 partial loss sums

  auto conv = [&](const float* in, int wi, int bi, float* out,
                  int N, int Cin, int Hin, int Win, int Cout, int Hout,
                  int Wout, int KH, int KW, int stride, int pad) {
    // 8 waves / block, all sharing one 64-wide Cout tile
    int blocks = (Cout / 64) * ((N * Hout * Wout) / 128);
    if (KH == 3) {
      conv_wmma_kernel<3, 3><<<blocks, 256, 0, stream>>>(
          in, P(wi), P(bi), out, N, Cin, Hin, Win, Cout, Hout, Wout,
          stride, pad);
    } else {
      conv_wmma_kernel<1, 1><<<blocks, 256, 0, stream>>>(
          in, P(wi), P(bi), out, N, Cin, Hin, Win, Cout, Hout, Wout,
          stride, pad);
    }
  };
  auto gnorm = [&](const float* in, float* out, int gi, int bi, int C, int HW,
                   int dosilu) {
    groupnorm_kernel<<<8 * 32, 256, 0, stream>>>(in, out, P(gi), P(bi), C, HW,
                                                 dosilu);
  };
  auto vadd = [&](const float* a, const float* b, float* o, long n) {
    add_kernel<<<(int)((n + 255) / 256), 256, 0, stream>>>(a, b, o, (int)n);
  };
  auto resblock = [&](int c1b, int c1w, int c2b, int c2w, int n1b, int n1g,
                      int n2b, int n2g, int ninb, int ninw, int Cin, int Cout,
                      int H, int W) {
    int HW = H * W;
    gnorm(X, T1, n1g, n1b, Cin, HW, 1);
    conv(T1, c1w, c1b, T2, 8, Cin, H, W, Cout, H, W, 3, 3, 1, 1);
    gnorm(T2, T1, n2g, n2b, Cout, HW, 1);
    conv(T1, c2w, c2b, T2, 8, Cout, H, W, Cout, H, W, 3, 3, 1, 1);
    long n = 8L * Cout * HW;
    if (ninw >= 0) {
      conv(X, ninw, ninb, T1, 8, Cin, H, W, Cout, H, W, 1, 1, 1, 0);
      vadd(T1, T2, X, n);
    } else {
      vadd(X, T2, X, n);
    }
  };
  auto attnblock = [&](int kb, int kw, int nb, int ng, int pb, int pw, int qb,
                       int qw, int vb, int vw) {
    const int C = 512, H = 16, W = 16, L = 256;
    gnorm(X, T1, ng, nb, C, L, 0);
    conv(T1, qw, qb, Q,  8, C, H, W, C, H, W, 1, 1, 1, 0);
    conv(T1, kw, kb, Kt, 8, C, H, W, C, H, W, 1, 1, 1, 0);
    conv(T1, vw, vb, V,  8, C, H, W, C, H, W, 1, 1, 1, 0);
    { long waves = 8L * (L / 16) * (L / 16);
      attn_scores_kernel<<<(int)((waves + 7) / 8), 256, 0, stream>>>(
          Q, Kt, S, 8, C, L, 0.044194173824159216f); }   // 1/sqrt(512)
    softmax_kernel<<<8 * L, 256, 0, stream>>>(S, L);
    { long waves = 8L * (C / 16) * (L / 16);
      attn_out_kernel<<<(int)((waves + 7) / 8), 256, 0, stream>>>(V, S, T1, 8,
                                                                  C, L); }
    conv(T1, pw, pb, T2, 8, C, H, W, C, H, W, 1, 1, 1, 0);
    vadd(X, T2, X, 8L * C * L);
  };
  auto dsample = [&](int bi, int wi, int C, int H) {
    // reference: pad (0,1) bottom/right + stride-2 conv == OOB-as-zero here
    conv(X, wi, bi, T1, 8, C, H, H, C, H / 2, H / 2, 3, 3, 2, 0);
    float* t = X; X = T1; T1 = t;
  };

  // conv_in: 3 -> 128 @ 256  (x is leaf 147; conv_in b=1, w=2)
  conv(P(147), 2, 1, X, 8, 3, 256, 256, 128, 256, 256, 3, 3, 1, 1);
  // down[0] @256, 128->128
  resblock(5, 6, 7, 8, 9, 10, 11, 12, -1, -1, 128, 128, 256, 256);
  resblock(13, 14, 15, 16, 17, 18, 19, 20, -1, -1, 128, 128, 256, 256);
  dsample(21, 22, 128, 256);                                   // -> 128
  // down[1] @128, 128->128
  resblock(23, 24, 25, 26, 27, 28, 29, 30, -1, -1, 128, 128, 128, 128);
  resblock(31, 32, 33, 34, 35, 36, 37, 38, -1, -1, 128, 128, 128, 128);
  dsample(39, 40, 128, 128);                                   // -> 64
  // down[2] @64, 128->256 (res0 has nin: b=45, w=46)
  resblock(41, 42, 43, 44, 47, 48, 49, 50, 45, 46, 128, 256, 64, 64);
  resblock(51, 52, 53, 54, 55, 56, 57, 58, -1, -1, 256, 256, 64, 64);
  dsample(59, 60, 256, 64);                                    // -> 32
  // down[3] @32, 256->256
  resblock(61, 62, 63, 64, 65, 66, 67, 68, -1, -1, 256, 256, 32, 32);
  resblock(69, 70, 71, 72, 73, 74, 75, 76, -1, -1, 256, 256, 32, 32);
  dsample(77, 78, 256, 32);                                    // -> 16
  // down[4] @16, 256->512 with attention (res0 has nin: b=103, w=104)
  resblock(99, 100, 101, 102, 105, 106, 107, 108, 103, 104, 256, 512, 16, 16);
  attnblock(79, 80, 81, 82, 83, 84, 85, 86, 87, 88);
  resblock(109, 110, 111, 112, 113, 114, 115, 116, -1, -1, 512, 512, 16, 16);
  attnblock(89, 90, 91, 92, 93, 94, 95, 96, 97, 98);
  // mid
  resblock(127, 128, 129, 130, 131, 132, 133, 134, -1, -1, 512, 512, 16, 16);
  attnblock(117, 118, 119, 120, 121, 122, 123, 124, 125, 126);
  resblock(135, 136, 137, 138, 139, 140, 141, 142, -1, -1, 512, 512, 16, 16);
  // norm_out + silu -> conv_out (512->256, 3x3) -> quant_conv (1x1)
  gnorm(X, T1, 144, 143, 512, 256, 1);
  conv(T1, 4, 3, T2, 8, 512, 16, 16, 256, 16, 16, 3, 3, 1, 1);
  conv(T2, 146, 145, T1, 8, 256, 16, 16, 256, 16, 16, 1, 1, 1, 0);

  // vector quantization (codebook is leaf 0)
  cnorm_kernel<<<64, 256, 0, stream>>>(P(0), CN);
  vq_argmin_kernel<<<128, 32, 0, stream>>>(T1, P(0), CN, IDX);
  vq_finalize_kernel<<<2048, 256, 0, stream>>>(
      T1, P(0), IDX, (float*)d_out, LS, ((int*)d_out) + 524289);
  loss_reduce_kernel<<<1, 256, 0, stream>>>(LS, ((float*)d_out) + 524288);
}
